// ConvFlow_11871289606247
// MI455X (gfx1250) — compile-verified, matrix operand order checked by reading, not run on hardware
//
#include <hip/hip_runtime.h>
#include <hip/hip_bf16.h>
#include <hip/hip_fp16.h>
#include <math.h>

typedef __attribute__((ext_vector_type(16))) _Float16 v16h;
typedef __attribute__((ext_vector_type(8)))  float    v8f;

#define B_    16
#define T_    8192
#define F_    256
#define NL_   3
#define NB_   10
#define PROJ_ 29
#define TB_   5.0f
#define BT_   (B_ * T_)      // 131072 positions
#define RT_   (BT_ / 16)     // 8192 row tiles of 16
#define KC_   (F_ / 32)      // 8 K-chunks
#define NT_   (F_ / 16)      // 16 N-tiles

// ---------------------------------------------------------------------------
// K0: h = x0 @ w_pre + b_pre   (HALF=1, so it's an outer product per position)
// ---------------------------------------------------------------------------
__global__ void k_pre(const float* __restrict__ x, const float* __restrict__ w_pre,
                      const float* __restrict__ b_pre, float* __restrict__ h) {
  int idx = blockIdx.x * 256 + threadIdx.x;   // BT_*F_ threads
  int f  = idx & (F_ - 1);
  int bt = idx >> 8;
  h[idx] = x[bt * 2] * w_pre[f] + b_pre[f];
}

// ---------------------------------------------------------------------------
// K1: repack w11 (NL,F,F) f32 -> f16 WMMA B-fragment layout.
// Fragment (layer, kc, nt): lane L<16 holds col n=nt*16+L, K elems {0..7,16..23}+kc*32
//                           lane L>=16 holds col n=nt*16+L-16, K elems {8..15,24..31}+kc*32
// element e: k_local = e + (e>=8 ? 8 : 0) + (half ? 8 : 0)
// ---------------------------------------------------------------------------
__global__ void k_pack_w(const float* __restrict__ w11, _Float16* __restrict__ wp) {
  int idx   = blockIdx.x * 256 + threadIdx.x;  // NL_*KC_*NT_*512 = 196608
  int e     = idx & 15;
  int lane  = (idx >> 4) & 31;
  int nt    = (idx >> 9) & 15;
  int kc    = (idx >> 13) & 7;
  int layer = idx >> 16;
  int half  = lane >> 4;
  int n = nt * 16 + (lane & 15);
  int k = kc * 32 + e + (e >= 8 ? 8 : 0) + half * 8;
  wp[idx] = (_Float16)w11[(layer * F_ + k) * F_ + n];
}

// ---------------------------------------------------------------------------
// K2: depthwise dilated conv (k=3, SAME, zero pad) + LN1 + exact GELU,
// stored as f16 in WMMA A-fragment layout. One block (256 thr) per (b,t).
// A-fragment (rt, kc): lane L<16 -> row m=rt*16+L, K elems {0..7,16..23}
//                      lane L>=16 -> row m=rt*16+L-16, K elems {8..15,24..31}
// ---------------------------------------------------------------------------
__global__ void k_conv_ln1_gelu(const float* __restrict__ h, const float* __restrict__ mask,
                                const float* __restrict__ dw_k, const float* __restrict__ dw_b,
                                const float* __restrict__ g1, const float* __restrict__ b1,
                                int layer, int dil, _Float16* __restrict__ apack) {
  __shared__ float redS[8], redQ[8];
  int bt = blockIdx.x;
  int f  = threadIdx.x;
  int t  = bt & (T_ - 1);

  const float* kk = dw_k + layer * 3 * F_;     // dw_k[layer][w][0][f]
  float k0 = kk[f], k1 = kk[F_ + f], k2 = kk[2 * F_ + f];

  float c = h[bt * F_ + f] * mask[bt];
  float l = 0.f, r = 0.f;
  if (t >= dil)       l = h[(bt - dil) * F_ + f] * mask[bt - dil];
  if (t + dil < T_)   r = h[(bt + dil) * F_ + f] * mask[bt + dil];
  float y = l * k0 + c * k1 + r * k2 + dw_b[layer * F_ + f];

  // LayerNorm over 256 features
  float s = y, q = y * y;
  #pragma unroll
  for (int o = 16; o >= 1; o >>= 1) { s += __shfl_xor(s, o, 32); q += __shfl_xor(q, o, 32); }
  int wid = f >> 5;
  if ((f & 31) == 0) { redS[wid] = s; redQ[wid] = q; }
  __syncthreads();
  float ts = 0.f, tq = 0.f;
  #pragma unroll
  for (int i = 0; i < 8; ++i) { ts += redS[i]; tq += redQ[i]; }
  float mean = ts * (1.f / 256.f);
  float var  = tq * (1.f / 256.f) - mean * mean;
  float v  = (y - mean) * rsqrtf(var + 1e-5f) * g1[layer * F_ + f] + b1[layer * F_ + f];
  float ge = 0.5f * v * (1.f + erff(v * 0.70710678118f));  // exact gelu

  // scatter into A-fragment layout
  int rt = bt >> 4, mr = bt & 15;
  int kc = f >> 5,  kl = f & 31;
  int half = (kl >> 3) & 1;                 // kl in 8..15 or 24..31
  int e    = kl - (half ? 8 : 0) - (kl >= 16 ? 8 : 0);
  int lane2 = mr + 16 * half;
  apack[(rt * KC_ + kc) * 512 + lane2 * 16 + e] = (_Float16)ge;
}

// ---------------------------------------------------------------------------
// K3: y2 = A @ w11  via v_wmma_f32_16x16x32_f16, then +b11, LN2, GELU,
// residual h += y2.  One wave per 16-row tile x full N=256.
// 8 waves per block -> grid = RT_/8.  128 WMMAs per wave, accum in registers.
// ---------------------------------------------------------------------------
__global__ void __launch_bounds__(256)
k_gemm_ln2(const _Float16* __restrict__ apack, const _Float16* __restrict__ wp,
           const float* __restrict__ b11, const float* __restrict__ g2,
           const float* __restrict__ b2, int layer, float* __restrict__ h) {
  int wid  = threadIdx.x >> 5;
  int lane = threadIdx.x & 31;
  int rt   = blockIdx.x * 8 + wid;

  const _Float16* ap = apack + (size_t)rt * (KC_ * 512) + lane * 16;
  const _Float16* bp = wp + (size_t)layer * (KC_ * NT_ * 512) + lane * 16;

  v8f acc[NT_];
  #pragma unroll
  for (int nt = 0; nt < NT_; ++nt) acc[nt] = (v8f){0.f,0.f,0.f,0.f,0.f,0.f,0.f,0.f};

  #pragma unroll
  for (int kc = 0; kc < KC_; ++kc) {
    v16h af = *(const v16h*)(ap + kc * 512);
    #pragma unroll
    for (int nt = 0; nt < NT_; ++nt) {
      v16h bf = *(const v16h*)(bp + (kc * NT_ + nt) * 512);
      acc[nt] = __builtin_amdgcn_wmma_f32_16x16x32_f16(
          false, af, false, bf, (short)0, acc[nt], false, false);
    }
  }

  // Epilogue. C layout: VGPR r, lanes 0..15 -> (M=r, N=nt*16+lane);
  //                     lanes 16..31 -> (M=r+8, N=nt*16+lane-16).
  int nlo  = lane & 15;
  int half = lane >> 4;

  float bb[NT_], gg[NT_], lb[NT_];
  #pragma unroll
  for (int nt = 0; nt < NT_; ++nt) {
    int n = nt * 16 + nlo;
    bb[nt] = b11[layer * F_ + n];
    gg[nt] = g2 [layer * F_ + n];
    lb[nt] = b2 [layer * F_ + n];
  }
  #pragma unroll
  for (int nt = 0; nt < NT_; ++nt)
    #pragma unroll
    for (int r = 0; r < 8; ++r) acc[nt][r] += bb[nt];   // bias before LN stats

  #pragma unroll
  for (int r = 0; r < 8; ++r) {
    float s = 0.f, q = 0.f;
    #pragma unroll
    for (int nt = 0; nt < NT_; ++nt) { float v = acc[nt][r]; s += v; q += v * v; }
    // reduce within each 16-lane half (keeps the two rows separate)
    #pragma unroll
    for (int o = 1; o < 16; o <<= 1) { s += __shfl_xor(s, o, 32); q += __shfl_xor(q, o, 32); }
    float mean = s * (1.f / 256.f);
    float var  = q * (1.f / 256.f) - mean * mean;
    float rstd = rsqrtf(var + 1e-5f);
    int m = rt * 16 + r + half * 8;
    #pragma unroll
    for (int nt = 0; nt < NT_; ++nt) {
      int n = nt * 16 + nlo;
      float v  = (acc[nt][r] - mean) * rstd * gg[nt] + lb[nt];
      float ge = 0.5f * v * (1.f + erff(v * 0.70710678118f));
      h[(size_t)m * F_ + n] += ge;            // residual
    }
  }
}

// ---------------------------------------------------------------------------
// K4: final projection (256 -> 29) + rational-quadratic spline on x1.
// One wave per (b,t). Lane 0 evaluates the spline (K=10 bins).
// ---------------------------------------------------------------------------
__device__ __forceinline__ float softplusf(float x) {
  return (x > 20.f) ? x : log1pf(expf(x));
}

__global__ void k_proj_spline(const float* __restrict__ h, const float* __restrict__ x,
                              const float* __restrict__ mask,
                              const float* __restrict__ w_proj, const float* __restrict__ b_proj,
                              float* __restrict__ xo, float* __restrict__ lad_buf) {
  int bt = blockIdx.x;
  int lane = threadIdx.x;
  float msk = mask[bt];

  float hv[8];
  #pragma unroll
  for (int j = 0; j < 8; ++j) hv[j] = h[(size_t)bt * F_ + lane * 8 + j] * msk;

  __shared__ float pbuf[PROJ_];
  for (int p = 0; p < PROJ_; ++p) {
    float s = 0.f;
    #pragma unroll
    for (int j = 0; j < 8; ++j) s += hv[j] * w_proj[(lane * 8 + j) * PROJ_ + p];
    #pragma unroll
    for (int o = 16; o >= 1; o >>= 1) s += __shfl_xor(s, o, 32);
    if (lane == 0) pbuf[p] = (s + b_proj[p]) * msk;
  }
  __syncthreads();

  if (lane == 0) {
    const float scale = 1.f / 16.f;     // 1/sqrt(F)
    float x1 = x[bt * 2 + 1];

    // widths
    float uw[NB_]; float mw = -1e30f;
    #pragma unroll
    for (int k = 0; k < NB_; ++k) { uw[k] = pbuf[k] * scale; mw = fmaxf(mw, uw[k]); }
    float sw = 0.f;
    #pragma unroll
    for (int k = 0; k < NB_; ++k) { uw[k] = expf(uw[k] - mw); sw += uw[k]; }
    float cw[NB_ + 1]; cw[0] = 0.f;
    #pragma unroll
    for (int k = 0; k < NB_; ++k)
      cw[k + 1] = cw[k] + (0.001f + (1.f - 0.001f * NB_) * (uw[k] / sw));
    #pragma unroll
    for (int k = 0; k <= NB_; ++k) cw[k] = 2.f * TB_ * cw[k] - TB_;
    cw[0] = -TB_; cw[NB_] = TB_;

    // heights
    float uh[NB_]; float mh = -1e30f;
    #pragma unroll
    for (int k = 0; k < NB_; ++k) { uh[k] = pbuf[NB_ + k] * scale; mh = fmaxf(mh, uh[k]); }
    float sh = 0.f;
    #pragma unroll
    for (int k = 0; k < NB_; ++k) { uh[k] = expf(uh[k] - mh); sh += uh[k]; }
    float ch[NB_ + 1]; ch[0] = 0.f;
    #pragma unroll
    for (int k = 0; k < NB_; ++k)
      ch[k + 1] = ch[k] + (0.001f + (1.f - 0.001f * NB_) * (uh[k] / sh));
    #pragma unroll
    for (int k = 0; k <= NB_; ++k) ch[k] = 2.f * TB_ * ch[k] - TB_;
    ch[0] = -TB_; ch[NB_] = TB_;

    // derivatives (9 inner, padded both ends with const)
    float dconst = logf(expm1f(1.f - 0.001f));
    float d[NB_ + 1];
    d[0]   = 0.001f + softplusf(dconst);
    #pragma unroll
    for (int k = 1; k < NB_; ++k) d[k] = 0.001f + softplusf(pbuf[2 * NB_ + k - 1]);
    d[NB_] = 0.001f + softplusf(dconst);

    bool inside = (x1 >= -TB_) && (x1 <= TB_);
    float xc = fminf(fmaxf(x1, -TB_), TB_);
    int idx = -1;
    #pragma unroll
    for (int k = 0; k <= NB_; ++k) idx += (xc >= cw[k]) ? 1 : 0;
    idx = min(max(idx, 0), NB_ - 1);

    float icw = cw[idx], ibw = cw[idx + 1] - cw[idx];
    float ich = ch[idx], ibh = ch[idx + 1] - ch[idx];
    float delta = ibh / ibw;
    float dk = d[idx], dk1 = d[idx + 1];
    float th   = (xc - icw) / ibw;
    float th1m = th * (1.f - th);
    float denom = delta + (dk + dk1 - 2.f * delta) * th1m;
    float num   = ibh * (delta * th * th + dk * th1m);
    float yv    = ich + num / denom;
    float omth  = 1.f - th;
    float dnum  = delta * delta * (dk1 * th * th + 2.f * delta * th1m + dk * omth * omth);
    float lad   = logf(dnum) - 2.f * logf(denom);

    float y1 = inside ? yv : x1;
    lad      = inside ? lad : 0.f;

    xo[bt * 2 + 0] = x[bt * 2 + 0] * msk;
    xo[bt * 2 + 1] = y1 * msk;
    lad_buf[bt]    = lad * msk;
  }
}

// ---------------------------------------------------------------------------
// K5: logdet[b] = sum_t lad[b,t]  (deterministic tree reduction)
// ---------------------------------------------------------------------------
__global__ void k_logdet(const float* __restrict__ lad, float* __restrict__ out_logdet) {
  __shared__ float red[8];
  int b = blockIdx.x, tid = threadIdx.x;
  float s = 0.f;
  for (int t = tid; t < T_; t += 256) s += lad[b * T_ + t];
  #pragma unroll
  for (int o = 16; o >= 1; o >>= 1) s += __shfl_xor(s, o, 32);
  if ((tid & 31) == 0) red[tid >> 5] = s;
  __syncthreads();
  if (tid == 0) {
    float t = 0.f;
    #pragma unroll
    for (int i = 0; i < 8; ++i) t += red[i];
    out_logdet[b] = t;
  }
}

// ---------------------------------------------------------------------------
extern "C" void kernel_launch(void* const* d_in, const int* in_sizes, int n_in,
                              void* d_out, int out_size, void* d_ws, size_t ws_size,
                              hipStream_t stream) {
  const float* x      = (const float*)d_in[0];
  const float* mask   = (const float*)d_in[1];
  const float* w_pre  = (const float*)d_in[2];
  const float* b_pre  = (const float*)d_in[3];
  const float* dw_k   = (const float*)d_in[4];
  const float* dw_b   = (const float*)d_in[5];
  const float* w11    = (const float*)d_in[6];
  const float* b11    = (const float*)d_in[7];
  const float* ln1_g  = (const float*)d_in[8];
  const float* ln1_b  = (const float*)d_in[9];
  const float* ln2_g  = (const float*)d_in[10];
  const float* ln2_b  = (const float*)d_in[11];
  const float* w_proj = (const float*)d_in[12];
  const float* b_proj = (const float*)d_in[13];
  float* out = (float*)d_out;

  char* ws = (char*)d_ws;
  const size_t off_h     = 0;
  const size_t off_apack = off_h + (size_t)BT_ * F_ * sizeof(float);       // 134,217,728
  const size_t off_wpack = off_apack + (size_t)BT_ * F_ * sizeof(_Float16); // +67,108,864
  const size_t off_lad   = off_wpack + (size_t)NL_ * F_ * F_ * sizeof(_Float16); // +393,216
  float*     h     = (float*)(ws + off_h);
  _Float16*  apack = (_Float16*)(ws + off_apack);
  _Float16*  wpack = (_Float16*)(ws + off_wpack);
  float*     ladb  = (float*)(ws + off_lad);

  k_pre<<<BT_, 256, 0, stream>>>(x, w_pre, b_pre, h);
  k_pack_w<<<(NL_ * KC_ * NT_ * 512) / 256, 256, 0, stream>>>(w11, wpack);

  int dil = 1;
  for (int i = 0; i < NL_; ++i) {
    k_conv_ln1_gelu<<<BT_, 256, 0, stream>>>(h, mask, dw_k, dw_b, ln1_g, ln1_b, i, dil, apack);
    k_gemm_ln2<<<RT_ / 8, 256, 0, stream>>>(apack, wpack, b11, ln2_g, ln2_b, i, h);
    dil *= 3;
  }

  k_proj_spline<<<BT_, 32, 0, stream>>>(h, x, mask, w_proj, b_proj, out, ladb);
  k_logdet<<<B_, 256, 0, stream>>>(ladb, out + (size_t)BT_ * 2);
}
